// MlpExtractor_66254165508823
// MI455X (gfx1250) — matrix operand order repaired and checked by reading, not compiled, not run on hardware
//
#include <hip/hip_runtime.h>
#include <hip/hip_bf16.h>

// ---------------------------------------------------------------------------
// Fused MoE MLP extractor for gfx1250 (MI455X), wave32 + WMMA f16->f32.
// Streaming (features/outputs) uses non-temporal hints to preserve L2 for the
// ~100 KB weight-fragment working set shared by all workgroups.
// ---------------------------------------------------------------------------

typedef _Float16 h16;
typedef __attribute__((ext_vector_type(16))) _Float16 v16h;
typedef __attribute__((ext_vector_type(4)))  _Float16 v4h;
typedef __attribute__((ext_vector_type(8)))  float    v8f;
typedef __attribute__((ext_vector_type(4)))  float    vf4;

#define B_ROWS 262144
#define WAVES  4
#define TILE_M 16

// workspace layout (in halves)
#define WS0T_OFF 0        // [128 n][128 k]
#define WS1T_OFF 16384    // [64  n][128 k]   (n pad 62->64 with 0)
#define WVT_OFF  24576    // [128 n][64  k]   (k pad 62->64 with 0)
#define WE1T_OFF 32768    // [2 e][64 n][64 k](k pad 62->64 with 0)
#define WE2T_OFF 40960    // [2 e][16 n][64 k](n pad 6->16 with 0)
#define WS_TOTAL 43008

union V16HU { vf4 f[2]; v16h h; };

// A-matrix 16-bit fragment (16x32), ISA 7.12.2:
// lanes 0-15 : K = kbase+{0..7, 16..23};  lanes 16-31: K = kbase+{8..15, 24..31}
__device__ __forceinline__ v16h afrag_lds(const h16* buf, int ks, int m, int kbase, int hi) {
  const h16* p = buf + m * ks + kbase + hi * 8;
  V16HU u;
  u.f[0] = *(const vf4*)(p);
  u.f[1] = *(const vf4*)(p + 16);
  return u.h;
}

// B-matrix 16-bit fragment (32x16): lane holds column n, same K order as A
// (lanes 0-15 -> K kbase+{0..7,16..23}, lanes 16-31 -> K kbase+{8..15,24..31}), from W^T[n][k].
__device__ __forceinline__ v16h bfrag_g(const h16* wt, int ks, int n, int kbase, int hi) {
  const h16* p = wt + n * ks + kbase + hi * 8;
  V16HU u;
  u.f[0] = *(const vf4*)(p);
  u.f[1] = *(const vf4*)(p + 16);
  return u.h;
}

__device__ __forceinline__ v8f wmma16(v16h a, v16h b, v8f c) {
  return __builtin_amdgcn_wmma_f32_16x16x32_f16(false, a, false, b, (short)0, c, false, false);
}

__device__ __forceinline__ float act_tanh(float x) {
#if __has_builtin(__builtin_amdgcn_tanhf)
  return __builtin_amdgcn_tanhf(x);   // V_TANH_F32 (gfx1250 TRANS op)
#else
  return tanhf(x);
#endif
}

// ---------------------------------------------------------------------------
// Prep: transpose + f16-convert + zero-pad all weight matrices into d_ws.
// ---------------------------------------------------------------------------
__global__ void prep_weights(const float* __restrict__ Ws0,
                             const float* __restrict__ Ws1,
                             const float* __restrict__ Wv,
                             const float* __restrict__ We1,
                             const float* __restrict__ We2,
                             h16* __restrict__ ws) {
  int idx = blockIdx.x * blockDim.x + threadIdx.x;
  if (idx < 16384) {                       // Ws0^T [128][128]
    int n = idx >> 7, k = idx & 127;
    ws[WS0T_OFF + idx] = (h16)Ws0[k * 128 + n];
  } else if (idx < 24576) {                // Ws1^T [64][128], n pad
    int i = idx - 16384;
    int n = i >> 7, k = i & 127;
    ws[WS1T_OFF + i] = (n < 62) ? (h16)Ws1[k * 62 + n] : (h16)0.0f;
  } else if (idx < 32768) {                // Wv^T [128][64], k pad
    int i = idx - 24576;
    int n = i >> 6, k = i & 63;
    ws[WVT_OFF + i] = (k < 62) ? (h16)Wv[k * 128 + n] : (h16)0.0f;
  } else if (idx < 40960) {                // We1^T [2][64][64], k pad
    int i = idx - 32768;
    int e = i >> 12, r = i & 4095;
    int n = r >> 6, k = r & 63;
    ws[WE1T_OFF + i] = (k < 62) ? (h16)We1[(e * 62 + k) * 64 + n] : (h16)0.0f;
  } else if (idx < WS_TOTAL) {             // We2^T [2][16][64], n pad
    int i = idx - 40960;
    int e = i >> 10, r = i & 1023;
    int n = r >> 6, k = r & 63;
    ws[WE2T_OFF + i] = (n < 6) ? (h16)We2[(e * 64 + k) * 6 + n] : (h16)0.0f;
  }
}

// ---------------------------------------------------------------------------
// Fused forward: each wave processes 16 rows end-to-end.
// ---------------------------------------------------------------------------
__global__ __launch_bounds__(WAVES * 32) void moe_fused(
    const float* __restrict__ feat, const float* __restrict__ noise,
    const float* __restrict__ bs0, const float* __restrict__ bs1,
    const float* __restrict__ bv,  const float* __restrict__ w_gate,
    const float* __restrict__ w_noise, const float* __restrict__ be1,
    const float* __restrict__ be2,
    const h16* __restrict__ ws0t, const h16* __restrict__ ws1t,
    const h16* __restrict__ wvt,  const h16* __restrict__ we1t,
    const h16* __restrict__ we2t,
    float* __restrict__ out) {
  // per-wave LDS slices
  __shared__ __align__(16) h16 sX[WAVES][TILE_M * 128];  // features, later expert-hidden
  __shared__ __align__(16) h16 sH[WAVES][TILE_M * 128];  // trunk hidden, later f32 logits
  __shared__ __align__(16) h16 sL[WAVES][TILE_M * 64];   // latent (padded to 64)
  __shared__ __align__(16) float sG[256];                // gating W^T: [gate|noise][2 e][64 k]

  const int w    = threadIdx.x >> 5;
  const int l    = threadIdx.x & 31;
  const int hi   = l >> 4;        // half-wave select
  const int nn   = l & 15;        // B/C column within 16-tile; also A row
  const int mb   = hi * 8;        // C row base for this half-wave
  const int rowbase = (blockIdx.x * WAVES + w) * TILE_M;

  h16* Xw = sX[w];
  h16* Hw = sH[w];
  h16* Lw = sL[w];

  // ---- stage 0a: stage gating weights transposed+padded into LDS (once/block)
  for (int i = threadIdx.x; i < 256; i += WAVES * 32) {
    int mat = i >> 7;            // 0 = gate, 1 = noise
    int e   = (i >> 6) & 1;
    int k   = i & 63;
    const float* src = mat ? w_noise : w_gate;
    sG[i] = (k < 62) ? src[k * 2 + e] : 0.0f;
  }

  // ---- stage 0b: features f32 -> f16 into LDS (coalesced NT b128 loads)
  #pragma unroll
  for (int i = 0; i < 16; ++i) {
    int idx = i * 32 + l;               // 512 groups of 4 floats
    int r = idx >> 5, g = idx & 31;
    vf4 f = __builtin_nontemporal_load(
        (const vf4*)(feat + (size_t)(rowbase + r) * 128 + g * 4));
    v4h hv;
    #pragma unroll
    for (int q = 0; q < 4; ++q) hv[q] = (h16)f[q];
    *(v4h*)(Xw + r * 128 + g * 4) = hv;
  }
  __syncthreads();

  // ---- stage 1: H = tanh(X @ Ws0 + bs0)   [16,128] x [128,128]
  {
    v16h a0 = afrag_lds(Xw, 128, nn, 0,  hi);
    v16h a1 = afrag_lds(Xw, 128, nn, 32, hi);
    v16h a2 = afrag_lds(Xw, 128, nn, 64, hi);
    v16h a3 = afrag_lds(Xw, 128, nn, 96, hi);
    #pragma unroll
    for (int nt = 0; nt < 8; ++nt) {
      int n = nt * 16 + nn;
      v8f c = {};
      c = wmma16(a0, bfrag_g(ws0t, 128, n, 0,  hi), c);
      c = wmma16(a1, bfrag_g(ws0t, 128, n, 32, hi), c);
      c = wmma16(a2, bfrag_g(ws0t, 128, n, 64, hi), c);
      c = wmma16(a3, bfrag_g(ws0t, 128, n, 96, hi), c);
      float b = bs0[n];
      #pragma unroll
      for (int r = 0; r < 8; ++r)
        Hw[(mb + r) * 128 + n] = (h16)act_tanh(c[r] + b);
    }
  }
  __syncthreads();

  // ---- stage 2: L = tanh(H @ Ws1 + bs1)   [16,128] x [128,64pad] (cols>=62 -> 0)
  {
    v16h a0 = afrag_lds(Hw, 128, nn, 0,  hi);
    v16h a1 = afrag_lds(Hw, 128, nn, 32, hi);
    v16h a2 = afrag_lds(Hw, 128, nn, 64, hi);
    v16h a3 = afrag_lds(Hw, 128, nn, 96, hi);
    #pragma unroll
    for (int nt = 0; nt < 4; ++nt) {
      int n = nt * 16 + nn;
      v8f c = {};
      c = wmma16(a0, bfrag_g(ws1t, 128, n, 0,  hi), c);
      c = wmma16(a1, bfrag_g(ws1t, 128, n, 32, hi), c);
      c = wmma16(a2, bfrag_g(ws1t, 128, n, 64, hi), c);
      c = wmma16(a3, bfrag_g(ws1t, 128, n, 96, hi), c);
      float b = (n < 62) ? bs1[n] : 0.0f;   // padded cols: w=0,b=0 -> tanh(0)=0
      #pragma unroll
      for (int r = 0; r < 8; ++r)
        Lw[(mb + r) * 64 + n] = (h16)act_tanh(c[r] + b);
    }
  }
  __syncthreads();

  // ---- stage 3: latent_vf = tanh(L @ Wv + bv)  [16,64] x [64,128], NT store
  {
    v16h a0 = afrag_lds(Lw, 64, nn, 0,  hi);
    v16h a1 = afrag_lds(Lw, 64, nn, 32, hi);
    float* vf_out = out + (size_t)B_ROWS * 6;
    #pragma unroll
    for (int nt = 0; nt < 8; ++nt) {
      int n = nt * 16 + nn;
      v8f c = {};
      c = wmma16(a0, bfrag_g(wvt, 64, n, 0,  hi), c);
      c = wmma16(a1, bfrag_g(wvt, 64, n, 32, hi), c);
      float b = bv[n];
      #pragma unroll
      for (int r = 0; r < 8; ++r)
        __builtin_nontemporal_store(
            act_tanh(c[r] + b),
            vf_out + (size_t)(rowbase + mb + r) * 128 + n);
    }
  }

  // ---- stage 4: EH[e] = relu(L @ We1[e] + be1[e])  -> reuse Xw as [16][2*64]
  {
    v16h a0 = afrag_lds(Lw, 64, nn, 0,  hi);
    v16h a1 = afrag_lds(Lw, 64, nn, 32, hi);
    #pragma unroll
    for (int e = 0; e < 2; ++e) {
      const h16* wt = we1t + e * 4096;
      #pragma unroll
      for (int nt = 0; nt < 4; ++nt) {
        int n = nt * 16 + nn;
        v8f c = {};
        c = wmma16(a0, bfrag_g(wt, 64, n, 0,  hi), c);
        c = wmma16(a1, bfrag_g(wt, 64, n, 32, hi), c);
        float b = be1[e * 64 + n];
        #pragma unroll
        for (int r = 0; r < 8; ++r)
          Xw[(mb + r) * 128 + e * 64 + n] = (h16)fmaxf(c[r] + b, 0.0f);
      }
    }
  }
  __syncthreads();

  // ---- stage 5: expert logits = EH @ We2 (+be2), N padded 6->16; f32 into Hw
  float* LOG = (float*)Hw;   // [16 rows][2 experts][8 pad] f32 (1 KB of 4 KB)
  {
    #pragma unroll
    for (int e = 0; e < 2; ++e) {
      v16h a0 = afrag_lds(Xw, 128, nn, e * 64,      hi);
      v16h a1 = afrag_lds(Xw, 128, nn, e * 64 + 32, hi);
      const h16* wt = we2t + e * 1024;
      v8f c = {};
      c = wmma16(a0, bfrag_g(wt, 64, nn, 0,  hi), c);
      c = wmma16(a1, bfrag_g(wt, 64, nn, 32, hi), c);
      if (nn < 6) {
        float b = be2[e * 6 + nn];
        #pragma unroll
        for (int r = 0; r < 8; ++r)
          LOG[(mb + r) * 16 + e * 8 + nn] = c[r] + b;
      }
    }
  }
  __syncthreads();

  // ---- stage 6: scalar tail — 2 lanes per row (lane = row*2 + expert)
  {
    const int row = l >> 1;
    const int e   = l & 1;
    const int grow = rowbase + row;

    // latent row -> registers (8x ds_load_b128); cols 62,63 are zero by padding
    h16 lv[64];
    #pragma unroll
    for (int c = 0; c < 4; ++c) {
      V16HU u;
      u.f[0] = *(const vf4*)(Lw + row * 64 + c * 16);
      u.f[1] = *(const vf4*)(Lw + row * 64 + c * 16 + 8);
      #pragma unroll
      for (int q = 0; q < 16; ++q) lv[c * 16 + q] = u.h[q];
    }

    // noisy gating dots against LDS-staged transposed weights (padded with 0)
    const float* wg = sG + e * 64;
    const float* wn = sG + 128 + e * 64;
    float cl = 0.0f, nz = 0.0f;
    #pragma unroll
    for (int k = 0; k < 64; ++k) {
      float x = (float)lv[k];
      cl += x * wg[k];
      nz += x * wn[k];
    }
    float sp    = fmaxf(nz, 0.0f) + log1pf(__expf(-fabsf(nz)));  // softplus
    float nsv   = __builtin_nontemporal_load(noise + (size_t)grow * 2 + e);
    float noisy = cl + nsv * (sp + 1e-2f);
    float other = __shfl_xor(noisy, 1, 32);
    float gmx   = fmaxf(noisy, other);
    float ge    = __expf(noisy - gmx);
    float gate  = ge / (ge + __expf(other - gmx));

    // expert softmax over 6 outputs
    float lg[6];
    float mx = -3.4e38f;
    #pragma unroll
    for (int o = 0; o < 6; ++o) {
      lg[o] = LOG[row * 16 + e * 8 + o];
      mx = fmaxf(mx, lg[o]);
    }
    float s = 0.0f;
    #pragma unroll
    for (int o = 0; o < 6; ++o) { lg[o] = __expf(lg[o] - mx); s += lg[o]; }
    float inv = 1.0f / s;

    float* eo_out = out + (size_t)B_ROWS * 134;   // after action(6) + vf(128)
    #pragma unroll
    for (int o = 0; o < 6; ++o) {
      lg[o] *= inv;
      __builtin_nontemporal_store(lg[o], eo_out + ((size_t)grow * 2 + e) * 6 + o);
    }

    // action = sum_e gate_e * expout_e  (pairwise reduce across lane pair)
    #pragma unroll
    for (int o = 0; o < 6; ++o) {
      float a = gate * lg[o];
      a += __shfl_xor(a, 1, 32);
      if (e == 0)
        __builtin_nontemporal_store(a, out + (size_t)grow * 6 + o);
    }
  }
}

// ---------------------------------------------------------------------------
extern "C" void kernel_launch(void* const* d_in, const int* in_sizes, int n_in,
                              void* d_out, int out_size, void* d_ws, size_t ws_size,
                              hipStream_t stream) {
  const float* feat    = (const float*)d_in[0];
  const float* noise   = (const float*)d_in[1];
  const float* Ws0     = (const float*)d_in[2];
  const float* bs0     = (const float*)d_in[3];
  const float* Ws1     = (const float*)d_in[4];
  const float* bs1     = (const float*)d_in[5];
  const float* Wv      = (const float*)d_in[6];
  const float* bv      = (const float*)d_in[7];
  const float* w_gate  = (const float*)d_in[8];
  const float* w_noise = (const float*)d_in[9];
  const float* We1     = (const float*)d_in[10];
  const float* be1     = (const float*)d_in[11];
  const float* We2     = (const float*)d_in[12];
  const float* be2     = (const float*)d_in[13];

  h16* wsh = (h16*)d_ws;

  prep_weights<<<(WS_TOTAL + 255) / 256, 256, 0, stream>>>(Ws0, Ws1, Wv, We1, We2, wsh);

  moe_fused<<<B_ROWS / (TILE_M * WAVES), WAVES * 32, 0, stream>>>(
      feat, noise, bs0, bs1, bv, w_gate, w_noise, be1, be2,
      wsh + WS0T_OFF, wsh + WS1T_OFF, wsh + WVT_OFF, wsh + WE1T_OFF, wsh + WE2T_OFF,
      (float*)d_out);
}
